// ModalityAxialBlock_1030792151579
// MI455X (gfx1250) — matrix operand order repaired and checked by reading, not compiled, not run on hardware
//
#include <hip/hip_runtime.h>
#include <hip/hip_bf16.h>

// ---------------------------------------------------------------------------
// Types
// ---------------------------------------------------------------------------
typedef __bf16  v16bf __attribute__((__ext_vector_type__(16)));
typedef float   f32x8 __attribute__((__ext_vector_type__(8)));
typedef unsigned int u32x4 __attribute__((__ext_vector_type__(4)));

union FragAB {
    v16bf v;
    u32x4 q[2];
};

__device__ __forceinline__ unsigned short f2bf(float f) {
    union { float f; unsigned u; } v; v.f = f;
    unsigned r = v.u + 0x7fffu + ((v.u >> 16) & 1u);   // round-to-nearest-even
    return (unsigned short)(r >> 16);
}
__device__ __forceinline__ float bf2f(unsigned short h) {
    union { unsigned u; float f; } v; v.u = ((unsigned)h) << 16; return v.f;
}

__device__ __forceinline__ f32x8 zero8() {
    f32x8 z = {0.f, 0.f, 0.f, 0.f, 0.f, 0.f, 0.f, 0.f};
    return z;
}

// LDS byte offset of a generic pointer to a __shared__ object (low 32 bits of
// the flat address are the LDS offset on gfx1250).
__device__ __forceinline__ unsigned lds_off(const void* p) {
    return (unsigned)(unsigned long long)(size_t)p;
}

// Async copy of 16B-per-lane chunks: global -> LDS, ASYNCcnt-tracked.
// IOFFSET is added to BOTH the LDS and the global address (ISA 15.18.3), so
// nbytes16 consecutive chunks share one base pair.
__device__ __forceinline__ void async_g2l_b128(unsigned lds, const void* gaddr) {
#if defined(__AMDGCN__)
    asm volatile("global_load_async_to_lds_b128 %0, %1, off"
                 :: "v"(lds), "v"(gaddr) : "memory");
#else
    (void)lds; (void)gaddr;
#endif
}
__device__ __forceinline__ void async_g2l_b128_off16(unsigned lds, const void* gaddr) {
#if defined(__AMDGCN__)
    asm volatile("global_load_async_to_lds_b128 %0, %1, off offset:16"
                 :: "v"(lds), "v"(gaddr) : "memory");
#else
    (void)lds; (void)gaddr;
#endif
}
__device__ __forceinline__ void async_g2l_b128_off32(unsigned lds, const void* gaddr) {
#if defined(__AMDGCN__)
    asm volatile("global_load_async_to_lds_b128 %0, %1, off offset:32"
                 :: "v"(lds), "v"(gaddr) : "memory");
#else
    (void)lds; (void)gaddr;
#endif
}
__device__ __forceinline__ void async_g2l_b128_off48(unsigned lds, const void* gaddr) {
#if defined(__AMDGCN__)
    asm volatile("global_load_async_to_lds_b128 %0, %1, off offset:48"
                 :: "v"(lds), "v"(gaddr) : "memory");
#else
    (void)lds; (void)gaddr;
#endif
}
__device__ __forceinline__ void wait_async0() {
#if defined(__AMDGCN__)
    asm volatile("s_wait_asynccnt 0x0" ::: "memory");
#endif
}

// A fragment (16x32 bf16): lane L: M = L%16, half-sel = L/16.
// halfs 0..7  -> K = sel*8 + 0..7 ; halfs 8..15 -> K = 16 + sel*8 + 0..7
__device__ __forceinline__ FragAB load_afrag(const unsigned short* sm, int row0,
                                             int stride, int koff) {
    int lane = threadIdx.x & 31;
    int sel  = lane >> 4;
    int r    = lane & 15;
    int base = (row0 + r) * stride + koff + sel * 8;
    FragAB f;
    f.q[0] = *reinterpret_cast<const u32x4*>(sm + base);
    f.q[1] = *reinterpret_cast<const u32x4*>(sm + base + 16);
    return f;
}

// B fragment (32x16 bf16), B stored transposed (Bt[n][k]): lane L: N = L%16,
// halfs 0..7 -> K = (L/16)*8 + 0..7 ; halfs 8..15 -> K = 16 + (L/16)*8 + 0..7
__device__ __forceinline__ FragAB load_bfrag(const unsigned short* sm, int row0,
                                             int stride, int koff) {
    int lane = threadIdx.x & 31;
    int sel  = lane >> 4;
    int n    = lane & 15;
    int base = (row0 + n) * stride + koff + sel * 8;
    FragAB f;
    f.q[0] = *reinterpret_cast<const u32x4*>(sm + base);
    f.q[1] = *reinterpret_cast<const u32x4*>(sm + base + 16);
    return f;
}

__device__ __forceinline__ f32x8 wmma_bf16(const FragAB& a, const FragAB& b, f32x8 c) {
    return __builtin_amdgcn_wmma_f32_16x16x32_bf16(false, a.v, false, b.v,
                                                   (short)0, c, false, false);
}

// ---------------------------------------------------------------------------
// Weight transpose + fp32->bf16 convert:  dst[c*R + r] = bf16(src[r*C + c])
// ---------------------------------------------------------------------------
__global__ void tconv_kernel(const float* __restrict__ src,
                             unsigned short* __restrict__ dst, int R, int C) {
    int idx = blockIdx.x * blockDim.x + threadIdx.x;
    if (idx >= R * C) return;
    int r = idx / C, c = idx % C;
    dst[(size_t)c * R + r] = f2bf(src[idx]);
}

// ---------------------------------------------------------------------------
// LayerNorm (per row of D), fp32 in -> bf16 out
// ---------------------------------------------------------------------------
__global__ __launch_bounds__(256) void ln_kernel(const float* __restrict__ x,
                                                 const float* __restrict__ scale,
                                                 const float* __restrict__ bias,
                                                 unsigned short* __restrict__ out,
                                                 int Dd) {
    __shared__ float s1[256];
    __shared__ float s2[256];
    int row = blockIdx.x;
    int tid = threadIdx.x;
    const float* xr = x + (size_t)row * Dd;
    float a = 0.f, b = 0.f;
    for (int c = tid; c < Dd; c += 256) { float v = xr[c]; a += v; b += v * v; }
    s1[tid] = a; s2[tid] = b;
    __syncthreads();
    for (int off = 128; off > 0; off >>= 1) {
        if (tid < off) { s1[tid] += s1[tid + off]; s2[tid] += s2[tid + off]; }
        __syncthreads();
    }
    float mean = s1[0] / (float)Dd;
    float var  = s2[0] / (float)Dd - mean * mean;
    float inv  = rsqrtf(var + 1e-6f);
    unsigned short* orow = out + (size_t)row * Dd;
    for (int c = tid; c < Dd; c += 256)
        orow[c] = f2bf((xr[c] - mean) * inv * scale[c] + bias[c]);
}

// ---------------------------------------------------------------------------
// Generic bf16 GEMM: C[M,N] = A[M,K] * Bt[N,K]^T (+bias, epilogue modes)
//   mode 0: out bf16 = acc + bias
//   mode 1: out bf16 = gelu(acc + bias)
//   mode 2: out f32  = res + acc + bias
// Tile 128x128x64, 256 threads (8 waves, each 64x32 of output).
// Staging uses GLOBAL_LOAD_ASYNC_TO_LDS_B128 (ASYNCcnt), 16 WMMA per stage.
// ---------------------------------------------------------------------------
#define BM 128
#define BN 128
#define BK 64
#define KSTR (BK + 8)   // 72 halves -> 144B row stride (16B aligned)

__global__ __launch_bounds__(256) void gemm_bf16(
    const unsigned short* __restrict__ A,
    const unsigned short* __restrict__ Bt,
    int M, int N, int K,
    const float* __restrict__ b0, const float* __restrict__ b1,
    const float* __restrict__ b2, int bseg, int mode,
    unsigned short* __restrict__ outb,
    const float* __restrict__ resf, float* __restrict__ outf) {
    __shared__ __align__(16) unsigned short As[BM * KSTR];
    __shared__ __align__(16) unsigned short Bs[BN * KSTR];
    const int m0   = blockIdx.y * BM;
    const int n0   = blockIdx.x * BN;
    const int tid  = threadIdx.x;
    const int wave = tid >> 5;
    const int wm   = wave & 1;   // 2 waves over M (64 rows each)
    const int wn   = wave >> 1;  // 4 waves over N (32 cols each)

    f32x8 acc[4][2];
#pragma unroll
    for (int i = 0; i < 4; ++i)
#pragma unroll
        for (int j = 0; j < 2; ++j) acc[i][j] = zero8();

    const int r  = tid >> 1;   // staging row 0..127
    const int ch = tid & 1;    // 64-byte half of the 128B row

    for (int k0 = 0; k0 < K; k0 += BK) {
        __syncthreads();
        {
            const unsigned short* gA = A + (size_t)(m0 + r) * K + k0 + ch * 32;
            unsigned lA = lds_off(As + r * KSTR) + ch * 64;
            async_g2l_b128(lA, gA);
            async_g2l_b128_off16(lA, gA);
            async_g2l_b128_off32(lA, gA);
            async_g2l_b128_off48(lA, gA);
            const unsigned short* gB = Bt + (size_t)(n0 + r) * K + k0 + ch * 32;
            unsigned lB = lds_off(Bs + r * KSTR) + ch * 64;
            async_g2l_b128(lB, gB);
            async_g2l_b128_off16(lB, gB);
            async_g2l_b128_off32(lB, gB);
            async_g2l_b128_off48(lB, gB);
            if (k0 + BK < K) {
                __builtin_prefetch(A + (size_t)(m0 + r) * K + k0 + BK, 0, 1);
                __builtin_prefetch(Bt + (size_t)(n0 + r) * K + k0 + BK, 0, 1);
            }
        }
        wait_async0();
        __syncthreads();
#pragma unroll
        for (int kc = 0; kc < BK; kc += 32) {
            FragAB bf0 = load_bfrag(Bs, wn * 32 + 0,  KSTR, kc);
            FragAB bf1 = load_bfrag(Bs, wn * 32 + 16, KSTR, kc);
#pragma unroll
            for (int ms = 0; ms < 4; ++ms) {
                FragAB af = load_afrag(As, wm * 64 + ms * 16, KSTR, kc);
                acc[ms][0] = wmma_bf16(af, bf0, acc[ms][0]);
                acc[ms][1] = wmma_bf16(af, bf1, acc[ms][1]);
            }
        }
    }

    const int lane = tid & 31, sel = lane >> 4, nl = lane & 15;
#pragma unroll
    for (int ms = 0; ms < 4; ++ms)
#pragma unroll
        for (int ns = 0; ns < 2; ++ns)
#pragma unroll
            for (int i = 0; i < 8; ++i) {
                int m = m0 + wm * 64 + ms * 16 + i + 8 * sel;
                int n = n0 + wn * 32 + ns * 16 + nl;
                float v = acc[ms][ns][i];
                int s = n / bseg;
                const float* bp = (s == 0) ? b0 : ((s == 1) ? b1 : b2);
                v += bp[n - s * bseg];
                size_t idx = (size_t)m * N + n;
                if (mode == 0) {
                    outb[idx] = f2bf(v);
                } else if (mode == 1) {
                    float t = v + 0.044715f * v * v * v;
                    float g = 0.5f * v * (1.0f + tanhf(0.7978845608028654f * t));
                    outb[idx] = f2bf(g);
                } else {
                    outf[idx] = resf[idx] + v;
                }
            }
}

// ---------------------------------------------------------------------------
// RoPE in-place on q,k halves of the packed qkv bf16 buffer (head_dim 64).
// ---------------------------------------------------------------------------
__global__ void rope_kernel(unsigned short* __restrict__ qkv, int NT_, int Mmod,
                            int Tlen) {
    int idx = blockIdx.x * blockDim.x + threadIdx.x;   // NT * H(16) * 32
    if (idx >= NT_ * 512) return;
    int row = idx >> 9;
    int hj  = idx & 511;
    int h   = hj >> 5;
    int j   = hj & 31;
    int t   = (row / Mmod) % Tlen;
    float inv = __expf(-(float)j * 0.28782313662425572f);  // ln(10000)/32
    float sn, cs;
    __sincosf((float)t * inv, &sn, &cs);
    size_t b0 = (size_t)row * 3072 + h * 64 + j;
    float q1 = bf2f(qkv[b0]),        q2 = bf2f(qkv[b0 + 32]);
    qkv[b0]        = f2bf(q1 * cs - q2 * sn);
    qkv[b0 + 32]   = f2bf(q2 * cs + q1 * sn);
    float k1 = bf2f(qkv[b0 + 1024]), k2 = bf2f(qkv[b0 + 1056]);
    qkv[b0 + 1024] = f2bf(k1 * cs - k2 * sn);
    qkv[b0 + 1056] = f2bf(k2 * cs + k1 * sn);
}

// ---------------------------------------------------------------------------
// Fused per-(sequence, head) attention. S = seq length (16 or 128), Dh = 64.
// One wave per 16 query rows (blockDim = (S/16)*32). Q/K staged via async-to-
// LDS; softmax is done in the WMMA C-fragment registers with cross-lane
// shuffles; P (bf16) aliases the dead Q/K LDS region, then O = P*V via WMMA.
// ---------------------------------------------------------------------------
#define DH 64
#define QKS (DH + 8)   // 72 halves per Q/K row

template <int S, int SK>
__global__ __launch_bounds__(256) void attn_kernel(
    const unsigned short* __restrict__ qkv,  // NT x 3072 (q|k|v)
    unsigned short* __restrict__ o,          // NT x 1024
    int H_, int den, int os, int is_, int rstride, int causal, float scale) {
    constexpr int NTILES = S / 16;
    constexpr int VTS    = SK + 8;
    constexpr int PS     = SK + 8;
    __shared__ __align__(16) unsigned short QK[2 * S * QKS];  // Q | K ; aliased by P
    __shared__ __align__(16) unsigned short Vt[DH * VTS];     // V^T [d][s], zero-pad

    const int bid  = blockIdx.x;
    const int h    = bid % H_;
    const int sid  = bid / H_;
    const int base = (sid / den) * os + (sid % den) * is_;
    const int tid  = threadIdx.x;
    const int nth  = blockDim.x;

    unsigned short* Qs = QK;
    unsigned short* Ks = QK + S * QKS;
    unsigned short* Pb = QK;  // alias after barrier

    // Stage Q, K via async-to-LDS (16-byte chunks; 8 chunks per 64-half row)
    for (int idx = tid; idx < S * 8; idx += nth) {
        int srow = idx >> 3, ch = idx & 7;
        const unsigned short* g =
            qkv + (size_t)(base + srow * rstride) * 3072 + h * 64 + ch * 8;
        async_g2l_b128(lds_off(Qs + srow * QKS + ch * 8), g);
        async_g2l_b128(lds_off(Ks + srow * QKS + ch * 8), g + 1024);
    }
    // Stage V transposed with zero pad up to SK
    for (int idx = tid; idx < DH * SK; idx += nth) {
        int d = idx / SK, s = idx % SK;
        unsigned short v = 0;
        if (s < S)
            v = qkv[(size_t)(base + s * rstride) * 3072 + 2048 + h * 64 + d];
        Vt[d * VTS + s] = v;
    }
    wait_async0();
    __syncthreads();

    const int wave = tid >> 5;         // == row tile (blockDim = NTILES*32)
    const int lane = tid & 31, sel = lane >> 4, nl = lane & 15;
    const int rt   = wave;

    // S = Q * K^T  (K-dim = Dh = 64 -> two wmma per tile)
    f32x8 sacc[NTILES];
#pragma unroll
    for (int nt = 0; nt < NTILES; ++nt) sacc[nt] = zero8();
    FragAB a0 = load_afrag(Qs, rt * 16, QKS, 0);
    FragAB a1 = load_afrag(Qs, rt * 16, QKS, 32);
#pragma unroll
    for (int nt = 0; nt < NTILES; ++nt) {
        FragAB b0 = load_bfrag(Ks, nt * 16, QKS, 0);
        FragAB b1 = load_bfrag(Ks, nt * 16, QKS, 32);
        sacc[nt] = wmma_bf16(a0, b0, sacc[nt]);
        sacc[nt] = wmma_bf16(a1, b1, sacc[nt]);
    }
    __syncthreads();  // everyone done reading Q/K before P overwrites them

    // scale + causal mask + softmax, in C-fragment registers
#pragma unroll
    for (int i = 0; i < 8; ++i) {
        int m = rt * 16 + i + 8 * sel;
        float mx = -3.0e38f;
#pragma unroll
        for (int nt = 0; nt < NTILES; ++nt) {
            int n = nt * 16 + nl;
            float v = sacc[nt][i] * scale;
            if (causal && n > m) v = -1.0e30f;
            sacc[nt][i] = v;
            mx = fmaxf(mx, v);
        }
#pragma unroll
        for (int d = 8; d >= 1; d >>= 1) mx = fmaxf(mx, __shfl_xor(mx, d, 32));
        float sum = 0.f;
#pragma unroll
        for (int nt = 0; nt < NTILES; ++nt) {
            float e = __expf(sacc[nt][i] - mx);
            sacc[nt][i] = e;
            sum += e;
        }
#pragma unroll
        for (int d = 8; d >= 1; d >>= 1) sum += __shfl_xor(sum, d, 32);
        float inv = 1.0f / sum;
#pragma unroll
        for (int nt = 0; nt < NTILES; ++nt) {
            int n = nt * 16 + nl;
            Pb[m * PS + n] = f2bf(sacc[nt][i] * inv);
        }
        if (SK > S) Pb[m * PS + S + nl] = 0;  // zero-pad K columns to 32
    }
    // intra-wave LDS store->load is in-order; each wave reads only its own
    // 16 P rows and the staged Vt -> no barrier needed here

    // O = P * V  (K-dim = SK)
    f32x8 oacc[4];
#pragma unroll
    for (int dt = 0; dt < 4; ++dt) oacc[dt] = zero8();
#pragma unroll
    for (int kc = 0; kc < SK; kc += 32) {
        FragAB af = load_afrag(Pb, rt * 16, PS, kc);
#pragma unroll
        for (int dt = 0; dt < 4; ++dt) {
            FragAB bf = load_bfrag(Vt, dt * 16, VTS, kc);
            oacc[dt] = wmma_bf16(af, bf, oacc[dt]);
        }
    }
#pragma unroll
    for (int dt = 0; dt < 4; ++dt)
#pragma unroll
        for (int i = 0; i < 8; ++i) {
            int m = rt * 16 + i + 8 * sel;
            int d = dt * 16 + nl;
            o[(size_t)(base + m * rstride) * 1024 + h * 64 + d] = f2bf(oacc[dt][i]);
        }
}

// ---------------------------------------------------------------------------
// Host orchestration
// ---------------------------------------------------------------------------
extern "C" void kernel_launch(void* const* d_in, const int* in_sizes, int n_in,
                              void* d_out, int out_size, void* d_ws, size_t ws_size,
                              hipStream_t stream) {
    (void)in_sizes; (void)n_in; (void)out_size; (void)ws_size;
    const int B = 8, T = 128, Mo = 16, H = 16, F = 4096;
    const int NT = B * T * Mo;  // 16384 tokens, D = 1024

    const float* x     = (const float*)d_in[0];
    const float* ln1_s = (const float*)d_in[1];
    const float* ln1_b = (const float*)d_in[2];
    const float* tWq = (const float*)d_in[3];  const float* tbq = (const float*)d_in[4];
    const float* tWk = (const float*)d_in[5];  const float* tbk = (const float*)d_in[6];
    const float* tWv = (const float*)d_in[7];  const float* tbv = (const float*)d_in[8];
    const float* tWo = (const float*)d_in[9];  const float* tbo = (const float*)d_in[10];
    const float* ln2_s = (const float*)d_in[11];
    const float* ln2_b = (const float*)d_in[12];
    const float* mWq = (const float*)d_in[13]; const float* mbq = (const float*)d_in[14];
    const float* mWk = (const float*)d_in[15]; const float* mbk = (const float*)d_in[16];
    const float* mWv = (const float*)d_in[17]; const float* mbv = (const float*)d_in[18];
    const float* mWo = (const float*)d_in[19]; const float* mbo = (const float*)d_in[20];
    const float* ln3_s = (const float*)d_in[21];
    const float* ln3_b = (const float*)d_in[22];
    const float* W1 = (const float*)d_in[23];  const float* b1 = (const float*)d_in[24];
    const float* W2 = (const float*)d_in[25];  const float* b2 = (const float*)d_in[26];

    // workspace carve-up (all sizes are multiples of 256B)
    char* ws = (char*)d_ws;
    size_t off = 0;
    auto carve = [&](size_t bytes) {
        void* p = ws + off;
        off += (bytes + 255) & ~(size_t)255;
        return p;
    };
    unsigned short* wqkv1 = (unsigned short*)carve(3ull * 1024 * 1024 * 2);
    unsigned short* wo1   = (unsigned short*)carve(1ull * 1024 * 1024 * 2);
    unsigned short* wqkv2 = (unsigned short*)carve(3ull * 1024 * 1024 * 2);
    unsigned short* wo2   = (unsigned short*)carve(1ull * 1024 * 1024 * 2);
    unsigned short* w1t   = (unsigned short*)carve((size_t)F * 1024 * 2);
    unsigned short* w2t   = (unsigned short*)carve((size_t)F * 1024 * 2);
    float*          xbuf  = (float*)carve((size_t)NT * 1024 * 4);
    unsigned short* hbuf  = (unsigned short*)carve((size_t)NT * 1024 * 2);
    unsigned short* qkv   = (unsigned short*)carve((size_t)NT * 3072 * 2);
    unsigned short* obuf  = (unsigned short*)carve((size_t)NT * 1024 * 2);
    unsigned short* hid   = qkv;  // MLP hidden (NT x 4096) overlays qkv+obuf

    hipMemcpyAsync(xbuf, x, (size_t)NT * 1024 * 4, hipMemcpyDeviceToDevice, stream);

    auto tc = [&](const float* s, unsigned short* d, int R, int C) {
        int n = R * C;
        tconv_kernel<<<(n + 255) / 256, 256, 0, stream>>>(s, d, R, C);
    };
    tc(tWq, wqkv1 + 0,               1024, 1024);
    tc(tWk, wqkv1 + 1024 * 1024,     1024, 1024);
    tc(tWv, wqkv1 + 2 * 1024 * 1024, 1024, 1024);
    tc(tWo, wo1, 1024, 1024);
    tc(mWq, wqkv2 + 0,               1024, 1024);
    tc(mWk, wqkv2 + 1024 * 1024,     1024, 1024);
    tc(mWv, wqkv2 + 2 * 1024 * 1024, 1024, 1024);
    tc(mWo, wo2, 1024, 1024);
    tc(W1, w1t, 1024, 4096);
    tc(W2, w2t, 4096, 1024);

    // ----- block 1: temporal causal attention with RoPE -----
    ln_kernel<<<NT, 256, 0, stream>>>(xbuf, ln1_s, ln1_b, hbuf, 1024);
    gemm_bf16<<<dim3(3072 / BN, NT / BM), 256, 0, stream>>>(
        hbuf, wqkv1, NT, 3072, 1024, tbq, tbk, tbv, 1024, 0, qkv, nullptr, nullptr);
    rope_kernel<<<(NT * 512 + 255) / 256, 256, 0, stream>>>(qkv, NT, Mo, T);
    // seq over t: base = b*T*M + m, stride = M ; grid = B*M*H
    attn_kernel<128, 128><<<B * Mo * H, 256, 0, stream>>>(
        qkv, obuf, H, Mo, T * Mo, 1, Mo, 1, 0.125f);
    gemm_bf16<<<dim3(1024 / BN, NT / BM), 256, 0, stream>>>(
        obuf, wo1, NT, 1024, 1024, tbo, tbo, tbo, 1024, 2, nullptr, xbuf, xbuf);

    // ----- block 2: modality attention (S = 16, no mask, no rope) -----
    ln_kernel<<<NT, 256, 0, stream>>>(xbuf, ln2_s, ln2_b, hbuf, 1024);
    gemm_bf16<<<dim3(3072 / BN, NT / BM), 256, 0, stream>>>(
        hbuf, wqkv2, NT, 3072, 1024, mbq, mbk, mbv, 1024, 0, qkv, nullptr, nullptr);
    // seq over m: base = (b*T+t)*M, stride = 1 ; grid = B*T*H, 1 wave/block
    attn_kernel<16, 32><<<B * T * H, 32, 0, stream>>>(
        qkv, obuf, H, B * T, 0, Mo, 1, 0, 0.125f);
    gemm_bf16<<<dim3(1024 / BN, NT / BM), 256, 0, stream>>>(
        obuf, wo2, NT, 1024, 1024, mbo, mbo, mbo, 1024, 2, nullptr, xbuf, xbuf);

    // ----- MLP -----
    ln_kernel<<<NT, 256, 0, stream>>>(xbuf, ln3_s, ln3_b, hbuf, 1024);
    gemm_bf16<<<dim3(4096 / BN, NT / BM), 256, 0, stream>>>(
        hbuf, w1t, NT, 4096, 1024, b1, b1, b1, 4096, 1, hid, nullptr, nullptr);
    gemm_bf16<<<dim3(1024 / BN, NT / BM), 256, 0, stream>>>(
        hid, w2t, NT, 1024, 4096, b2, b2, b2, 1024, 2, nullptr, xbuf, (float*)d_out);
}
